// GlobalGraph_12627203851069
// MI455X (gfx1250) — matrix operand order repaired and checked by reading, not compiled, not run on hardware
//
#include <hip/hip_runtime.h>
#include <hip/hip_bf16.h>

// ---------------------------------------------------------------------------
// Fused attention for MI455X (gfx1250): bf16 WMMA 16x16x32, fp32 accumulate.
// B=16, S=1024, HID=512, H=8, D=64.
// Round 2: register-blocked (32x32 GEMM tiles, 32 queries/wave attention)
// to double operand reuse per WMMA.
// ---------------------------------------------------------------------------

typedef __bf16 bf16_t;
typedef bf16_t v16bf __attribute__((ext_vector_type(16)));
typedef bf16_t v8bf  __attribute__((ext_vector_type(8)));
typedef float  v8f   __attribute__((ext_vector_type(8)));

#define BATCH 16
#define SEQ   1024
#define HID   512
#define NH    8
#define DH    64

static __device__ __forceinline__ v8f wmma_bf16(v16bf a, v16bf b, v8f c) {
  // (neg_a, A, neg_b, B, c_mod, C, reuse_a, reuse_b)
  return __builtin_amdgcn_wmma_f32_16x16x32_bf16(false, a, false, b, (short)0, c,
                                                 false, false);
}

static __device__ __forceinline__ v16bf load_a_op(const bf16_t* row, int kk, int hi) {
  // A-operand (16x32, M x K): lane m = lane&15; e -> k=kk+16*hi+e.
  // One 32B contiguous load.
  v16bf a;
  a = *(const v16bf*)(row + kk + 16 * hi);
  return a;
}

// ------------------------------ conversion ---------------------------------
__global__ void f32_to_bf16_kernel(const float* __restrict__ src,
                                   bf16_t* __restrict__ dst, int n) {
  int i = blockIdx.x * blockDim.x + threadIdx.x;
  int stride = gridDim.x * blockDim.x;
  for (; i < n; i += stride) dst[i] = (bf16_t)src[i];
}

// ------------------------------ QKV GEMM -----------------------------------
// z = 0:Q (bias bq, scale 1/8), 1:K (no bias), 2:V (bias bv, stored transposed)
// One wave computes a 32x32 tile: 2 A-tiles x 2 B-tiles, 4 WMMAs per k-step
// sharing operands (16 FLOP per cache byte vs 8 for 16x16 tiles).
__global__ __launch_bounds__(128) void qkv_gemm_kernel(
    const bf16_t* __restrict__ X,   // [B*S, HID] bf16
    const bf16_t* __restrict__ Wq, const bf16_t* __restrict__ Wk,
    const bf16_t* __restrict__ Wv,  // each [HID, HID] bf16 (row = o)
    const float* __restrict__ bq, const float* __restrict__ bv,
    bf16_t* __restrict__ Qb,        // [B*S, HID] bf16 (pre-scaled by 1/8)
    bf16_t* __restrict__ Kb,        // [B*S, HID] bf16
    bf16_t* __restrict__ Vt)        // [B*NH, DH, SEQ] bf16 (transposed)
{
  const int lane = threadIdx.x & 31;
  const int wave = threadIdx.x >> 5;
  const int n16  = lane & 15;   // tile-local n (and A-row m)
  const int hi   = lane >> 4;   // half select

  const int mb  = blockIdx.x * 32;
  const int nbw = blockIdx.y * 128 + wave * 32;
  const int z   = blockIdx.z;

  const bf16_t* W = (z == 0) ? Wq : ((z == 1) ? Wk : Wv);

  const bf16_t* xrow0 = X + (size_t)(mb + n16) * HID;
  const bf16_t* xrow1 = X + (size_t)(mb + 16 + n16) * HID;
  const bf16_t* wrow0 = W + (size_t)(nbw + n16) * HID;
  const bf16_t* wrow1 = W + (size_t)(nbw + 16 + n16) * HID;

  v8f acc00 = {}, acc01 = {}, acc10 = {}, acc11 = {};
#pragma unroll
  for (int kk = 0; kk < HID; kk += 32) {
    v16bf a0 = load_a_op(xrow0, kk, hi);
    v16bf a1 = load_a_op(xrow1, kk, hi);
    // B-operand (32x16, K x N): lane n = lane&15; halves e -> k = kk+16*hi+e.
    v16bf b0 = *(const v16bf*)(wrow0 + kk + 16 * hi);
    v16bf b1 = *(const v16bf*)(wrow1 + kk + 16 * hi);
    acc00 = wmma_bf16(a0, b0, acc00);
    acc01 = wmma_bf16(a0, b1, acc01);
    acc10 = wmma_bf16(a1, b0, acc10);
    acc11 = wmma_bf16(a1, b1, acc11);
  }

  float bias0 = 0.0f, bias1 = 0.0f;
  if (z == 0) { bias0 = bq[nbw + n16]; bias1 = bq[nbw + 16 + n16]; }
  else if (z == 2) { bias0 = bv[nbw + n16]; bias1 = bv[nbw + 16 + n16]; }
  const float scale = (z == 0) ? 0.125f : 1.0f;  // fold 1/sqrt(D) into Q

  if (z < 2) {
    bf16_t* dst = (z == 0) ? Qb : Kb;
#pragma unroll
    for (int r = 0; r < 8; ++r) {
      int m0 = mb + r + 8 * hi;
      int m1 = mb + 16 + r + 8 * hi;
      dst[(size_t)m0 * HID + nbw + n16]      = (bf16_t)((acc00[r] + bias0) * scale);
      dst[(size_t)m0 * HID + nbw + 16 + n16] = (bf16_t)((acc01[r] + bias1) * scale);
      dst[(size_t)m1 * HID + nbw + n16]      = (bf16_t)((acc10[r] + bias0) * scale);
      dst[(size_t)m1 * HID + nbw + 16 + n16] = (bf16_t)((acc11[r] + bias1) * scale);
    }
  } else {
#pragma unroll
    for (int mt = 0; mt < 2; ++mt) {
#pragma unroll
      for (int nt = 0; nt < 2; ++nt) {
#pragma unroll
        for (int r = 0; r < 8; ++r) {
          float v = (mt == 0) ? ((nt == 0) ? acc00[r] : acc01[r])
                              : ((nt == 0) ? acc10[r] : acc11[r]);
          float bsv = (nt == 0) ? bias0 : bias1;
          int m = mb + mt * 16 + r + 8 * hi;
          int b = m >> 10;            // m / SEQ
          int s = m & (SEQ - 1);
          int o = nbw + nt * 16 + n16;  // = h*64 + d
          int h = o >> 6, d = o & 63;
          Vt[((size_t)(b * NH + h) * DH + d) * SEQ + s] = (bf16_t)(v + bsv);
        }
      }
    }
  }
}

// ------------------------------ attention ----------------------------------
// Block = 4 waves (128 thr); each wave = 32 queries (2 query tiles) of one
// (b,h). K-tile A-operands and Vt B-operands are reused across both query
// tiles: 16 WMMAs per 32-key chunk against 24 b128 loads per lane.
// scores^T via WMMA => lane = query i, VGPR r = key j. Softmax rows reduce
// with 8 register ops + shfl_xor(16); P lands directly in A-operand layout.
__global__ __launch_bounds__(128) void attn_kernel(
    const bf16_t* __restrict__ Qb,   // [B*S, HID] (pre-scaled)
    const bf16_t* __restrict__ Kb,   // [B*S, HID]
    const bf16_t* __restrict__ Vt,   // [B*NH, DH, SEQ]
    const float*  __restrict__ mask, // [B, S, S]
    float* __restrict__ out)         // [B, S, HID]
{
  const int lane = threadIdx.x & 31;
  const int wave = threadIdx.x >> 5;
  const int n16  = lane & 15;
  const int hi   = lane >> 4;

  const int b = blockIdx.z;
  const int h = blockIdx.y;
  const int ibase = blockIdx.x * 128 + wave * 32;

  // Q as B-operand: lane n = query i; halves e -> d = kk + 16*hi + e
  const bf16_t* qrow0 = Qb + ((size_t)(b * SEQ + ibase + n16)) * HID + h * DH;
  const bf16_t* qrow1 = Qb + ((size_t)(b * SEQ + ibase + 16 + n16)) * HID + h * DH;
  v16bf qb00 = *(const v16bf*)(qrow0 + 16 * hi);       // qt0, d in [0,32)
  v16bf qb01 = *(const v16bf*)(qrow0 + 32 + 16 * hi);  // qt0, d in [32,64)
  v16bf qb10 = *(const v16bf*)(qrow1 + 16 * hi);       // qt1
  v16bf qb11 = *(const v16bf*)(qrow1 + 32 + 16 * hi);

  const bf16_t* Kbase = Kb + ((size_t)b * SEQ) * HID + h * DH;
  const bf16_t* Vbase = Vt + ((size_t)(b * NH + h) * DH) * SEQ;
  const float*  mrowp0 = mask + ((size_t)b * SEQ + ibase + n16) * SEQ;
  const float*  mrowp1 = mask + ((size_t)b * SEQ + ibase + 16 + n16) * SEQ;

  v8f acc[2][4] = {};
  float mrow[2] = {-INFINITY, -INFINITY};
  float lrow[2] = {0.0f, 0.0f};
  float p0[2][8], p1[2][8];  // exp(scores) per query tile / key subtile

  for (int jb = 0; jb < SEQ; jb += 32) {
    // ---- K A-operands (shared by both query tiles) ----
    const bf16_t* krow0 = Kbase + (size_t)(jb + n16) * HID;
    const bf16_t* krow1 = Kbase + (size_t)(jb + 16 + n16) * HID;
    v16bf ka00 = load_a_op(krow0, 0, hi);   // subtile0, d [0,32)
    v16bf ka01 = load_a_op(krow0, 32, hi);  // subtile0, d [32,64)
    v16bf ka10 = load_a_op(krow1, 0, hi);   // subtile1
    v16bf ka11 = load_a_op(krow1, 32, hi);

    // ---- scores^T: 8 WMMAs (2 key subtiles x 2 d-chunks x 2 query tiles) --
    v8f sv[2][2];  // [key subtile][query tile]
    {
      v8f s;
      s = wmma_bf16(ka00, qb00, (v8f){}); sv[0][0] = wmma_bf16(ka01, qb01, s);
      s = wmma_bf16(ka00, qb10, (v8f){}); sv[0][1] = wmma_bf16(ka01, qb11, s);
      s = wmma_bf16(ka10, qb00, (v8f){}); sv[1][0] = wmma_bf16(ka11, qb01, s);
      s = wmma_bf16(ka10, qb10, (v8f){}); sv[1][1] = wmma_bf16(ka11, qb11, s);
    }

    // ---- per query tile: mask + online softmax ----
#pragma unroll
    for (int qt = 0; qt < 2; ++qt) {
      const float* mrowp = (qt == 0) ? mrowp0 : mrowp1;
      float s0[8], s1[8], mv0[8], mv1[8];
      {
        const float* mp0 = mrowp + jb + 8 * hi;        // j = jb + r + 8*hi
        const float* mp1 = mrowp + jb + 16 + 8 * hi;   // j = jb+16 + r + 8*hi
        *(float4*)&mv0[0] = ((const float4*)mp0)[0];
        *(float4*)&mv0[4] = ((const float4*)mp0)[1];
        *(float4*)&mv1[0] = ((const float4*)mp1)[0];
        *(float4*)&mv1[4] = ((const float4*)mp1)[1];
      }
#pragma unroll
      for (int r = 0; r < 8; ++r) {
        s0[r] = sv[0][qt][r] + (1.0f - mv0[r]) * -10000.0f;
        s1[r] = sv[1][qt][r] + (1.0f - mv1[r]) * -10000.0f;
      }

      float cmax = s0[0];
#pragma unroll
      for (int r = 1; r < 8; ++r) cmax = fmaxf(cmax, s0[r]);
#pragma unroll
      for (int r = 0; r < 8; ++r) cmax = fmaxf(cmax, s1[r]);
      cmax = fmaxf(cmax, __shfl_xor(cmax, 16, 32));

      float mnew  = fmaxf(mrow[qt], cmax);
      float alpha = __expf(mrow[qt] - mnew);
      mrow[qt] = mnew;

      float psum = 0.0f;
#pragma unroll
      for (int r = 0; r < 8; ++r) { p0[qt][r] = __expf(s0[r] - mnew); psum += p0[qt][r]; }
#pragma unroll
      for (int r = 0; r < 8; ++r) { p1[qt][r] = __expf(s1[r] - mnew); psum += p1[qt][r]; }
      psum += __shfl_xor(psum, 16, 32);
      lrow[qt] = lrow[qt] * alpha + psum;

      // rescale accumulators: ctx row for VGPR r is i = r + 8*hi
      float alr[8];
#pragma unroll
      for (int r = 0; r < 8; ++r) alr[r] = __shfl(alpha, r + 8 * hi, 32);
#pragma unroll
      for (int r = 0; r < 8; ++r) {
        acc[qt][0][r] *= alr[r]; acc[qt][1][r] *= alr[r];
        acc[qt][2][r] *= alr[r]; acc[qt][3][r] *= alr[r];
      }
    }

    // ---- P in A-operand layout (lane m=i; e<8: k=8*hi+e from subtile0) ----
    v16bf pa0, pa1;
#pragma unroll
    for (int e = 0; e < 8; ++e) {
      pa0[e]     = (bf16_t)p0[0][e];
      pa0[e + 8] = (bf16_t)p1[0][e];
      pa1[e]     = (bf16_t)p0[1][e];
      pa1[e + 8] = (bf16_t)p1[1][e];
    }

    // ---- P*V: 4 shared Vt B-operands, 8 WMMAs ----
#pragma unroll
    for (int dt = 0; dt < 4; ++dt) {
      const bf16_t* vr = Vbase + (size_t)(dt * 16 + n16) * SEQ + jb;
      v16bf vb;
      ((v8bf*)&vb)[0] = *(const v8bf*)(vr + 8 * hi);
      ((v8bf*)&vb)[1] = *(const v8bf*)(vr + 16 + 8 * hi);
      acc[0][dt] = wmma_bf16(pa0, vb, acc[0][dt]);
      acc[1][dt] = wmma_bf16(pa1, vb, acc[1][dt]);
    }
  }

  // ---- epilogue: divide by row sum, store fp32 ----
#pragma unroll
  for (int qt = 0; qt < 2; ++qt) {
    float linv = 1.0f / lrow[qt];
    float lr[8];
#pragma unroll
    for (int r = 0; r < 8; ++r) lr[r] = __shfl(linv, r + 8 * hi, 32);
#pragma unroll
    for (int r = 0; r < 8; ++r) {
      size_t row = (size_t)(b * SEQ + ibase + qt * 16 + r + 8 * hi) * HID + h * DH + n16;
      out[row + 0]  = acc[qt][0][r] * lr[r];
      out[row + 16] = acc[qt][1][r] * lr[r];
      out[row + 32] = acc[qt][2][r] * lr[r];
      out[row + 48] = acc[qt][3][r] * lr[r];
    }
  }
}

// ------------------------------ launcher -----------------------------------
extern "C" void kernel_launch(void* const* d_in, const int* in_sizes, int n_in,
                              void* d_out, int out_size, void* d_ws, size_t ws_size,
                              hipStream_t stream) {
  const float* hs   = (const float*)d_in[0];  // [B,S,HID]
  const float* msk  = (const float*)d_in[1];  // [B,S,S]
  const float* Wq   = (const float*)d_in[2];
  const float* bq   = (const float*)d_in[3];
  const float* Wk   = (const float*)d_in[4];
  const float* Wv   = (const float*)d_in[5];
  const float* bv   = (const float*)d_in[6];
  float* out = (float*)d_out;

  const size_t nX = (size_t)BATCH * SEQ * HID;   // 8,388,608
  const size_t nW = (size_t)HID * HID;           // 262,144

  uint8_t* ws = (uint8_t*)d_ws;
  size_t off = 0;
  bf16_t* Xb  = (bf16_t*)(ws + off); off += nX * 2;   // 16 MiB
  bf16_t* Wqb = (bf16_t*)(ws + off); off += nW * 2;
  bf16_t* Wkb = (bf16_t*)(ws + off); off += nW * 2;
  bf16_t* Wvb = (bf16_t*)(ws + off); off += nW * 2;
  bf16_t* Qb  = (bf16_t*)(ws + off); off += nX * 2;
  bf16_t* Kb  = (bf16_t*)(ws + off); off += nX * 2;
  bf16_t* Vt  = (bf16_t*)(ws + off); off += nX * 2;   // total ~65.5 MiB

  // 1) fp32 -> bf16 conversions
  f32_to_bf16_kernel<<<2048, 256, 0, stream>>>(hs, Xb, (int)nX);
  f32_to_bf16_kernel<<<512, 256, 0, stream>>>(Wq, Wqb, (int)nW);
  f32_to_bf16_kernel<<<512, 256, 0, stream>>>(Wk, Wkb, (int)nW);
  f32_to_bf16_kernel<<<512, 256, 0, stream>>>(Wv, Wvb, (int)nW);

  // 2) QKV projections (Q pre-scaled by 1/sqrt(D); V stored transposed)
  dim3 ggrid(BATCH * SEQ / 32, HID / 128, 3);
  qkv_gemm_kernel<<<ggrid, 128, 0, stream>>>(Xb, Wqb, Wkb, Wvb, bq, bv,
                                             Qb, Kb, Vt);

  // 3) flash attention (128 queries per block, 32 per wave)
  dim3 agrid(SEQ / 128, NH, BATCH);
  attn_kernel<<<agrid, 128, 0, stream>>>(Qb, Kb, Vt, msk, out);
}
